// MyAttentionClassifier4_68186900791582
// MI455X (gfx1250) — matrix operand order repaired and checked by reading, not compiled, not run on hardware
//
#include <hip/hip_runtime.h>
#include <hip/hip_bf16.h>
#include <math.h>
#include <stdint.h>

typedef __bf16 bf16;
typedef __attribute__((ext_vector_type(16))) __bf16 v16bf;
typedef __attribute__((ext_vector_type(8)))  float  v8f;

#define WMMA_BF16(a, b, c) \
  __builtin_amdgcn_wmma_f32_16x16x32_bf16(false, (a), false, (b), (short)0, (c), false, false)

// ---------------------------------------------------------------- constants
#define EMBED   3072
#define HEADS   16
#define DH      192
#define QPROJ   1536
#define KVPROJ  2048
#define NOPE    96
#define ROPE    96
#define BATCH   8
#define SEQ     1024
#define ROWS    (BATCH * SEQ)          // 8192
#define KVW     (KVPROJ + ROPE)        // 2144
#define KVW_PAD 2176                   // padded to multiple of 64 for block tiling
#define UKVW    (EMBED + HEADS * NOPE) // 4608

union FragU { v16bf v; uint4 q[2]; };

// A-matrix fragment (16x32 bf16), contiguous k order used by this code:
// lanes 0-15: M=lane, K = 0..15; lanes 16-31: M=lane-16, K = 16..31
__device__ __forceinline__ v16bf load_a_frag(const bf16* A, int lda, int mbase,
                                             int kbase, int lane) {
  int m  = mbase + (lane & 15);
  int k0 = kbase + ((lane >> 4) << 4);
  const bf16* p = A + (size_t)m * lda + k0;
  FragU f;
  f.q[0] = *(const uint4*)(p);
  f.q[1] = *(const uint4*)(p + 8);
  return f.v;
}

// B-matrix fragment (32x16 bf16) from row-major B^T (N x K):
// lanes 0-15: N=lane, K=0..15; lanes 16-31: N=lane-16, K=16..31 (contiguous 32B per lane)
__device__ __forceinline__ v16bf load_b_frag(const bf16* Bt, int ldb, int nbase,
                                             int kbase, int lane) {
  int n  = nbase + (lane & 15);
  int kb = kbase + ((lane >> 4) << 4);
  const bf16* p = Bt + (size_t)n * ldb + kb;
  FragU f;
  f.q[0] = *(const uint4*)(p);
  f.q[1] = *(const uint4*)(p + 8);
  return f.v;
}

// CDNA5 async global->LDS copy, 16B per lane (ASYNCcnt tracked).
// lds_off: wave-relative LDS byte address == low 32 bits of generic &shared[i]
__device__ __forceinline__ void async_copy16(unsigned lds_off, const bf16* gaddr) {
  asm volatile("global_load_async_to_lds_b128 %0, %1, off"
               :: "v"(lds_off), "v"(gaddr) : "memory");
}
__device__ __forceinline__ void wait_asynccnt0() {
  asm volatile("s_wait_asynccnt 0" ::: "memory");
}

// ---------------------------------------------------------------- GEMM
// C(MxN,f32) = A(MxK,bf16 row-major) * B, with Bt = B^T (NxK row-major, rows
// padded so that reads up to ceil64(N) stay in-bounds).
// Block: 128 threads (4 waves), 64x64 block tile, each wave a 32x32 WMMA tile.
// A/B block tiles are double-buffered in LDS via global_load_async_to_lds_b128.
#define BM 64
#define BN 64
#define BK 32
__global__ __launch_bounds__(128) void gemm_bf16_wmma(
    const bf16* __restrict__ A, const bf16* __restrict__ Bt,
    float* __restrict__ C, int M, int N, int K) {
  __shared__ __align__(16) bf16 As[2][BM * BK];
  __shared__ __align__(16) bf16 Bs[2][BN * BK];
  const int tid  = threadIdx.x;
  const int lane = tid & 31;
  const int wave = tid >> 5;
  const int wm = (wave >> 1) * 32;   // wave M offset in block tile
  const int wn = (wave & 1) * 32;    // wave N offset in block tile
  const int mb = blockIdx.y * BM;
  const int nb = blockIdx.x * BN;

  // stage one 64x32 A tile + 64x32 B tile: 256 16B chunks each matrix half?
  // 64 rows * 4 segs = 256 chunks per matrix; thread t does chunks t, t+128.
  auto stage = [&](int kt, int buf) {
#pragma unroll
    for (int c = tid; c < 256; c += 128) {
      const int row = c >> 2, seg = (c & 3) * 8;  // seg in elements (8 bf16 = 16B)
      async_copy16((unsigned)(uintptr_t)&As[buf][row * BK + seg],
                   A + (size_t)(mb + row) * K + kt * BK + seg);
      async_copy16((unsigned)(uintptr_t)&Bs[buf][row * BK + seg],
                   Bt + (size_t)(nb + row) * K + kt * BK + seg);
    }
  };

  v8f acc00 = {0,0,0,0,0,0,0,0}, acc01 = {0,0,0,0,0,0,0,0};
  v8f acc10 = {0,0,0,0,0,0,0,0}, acc11 = {0,0,0,0,0,0,0,0};

  const int nk = K / BK;
  stage(0, 0);
  for (int kt = 0; kt < nk; ++kt) {
    const int buf = kt & 1;
    wait_asynccnt0();      // my staged chunks landed
    __syncthreads();       // everyone's staged chunks visible; prev reads done
    if (kt + 1 < nk) stage(kt + 1, buf ^ 1);
    const bf16* Al = As[buf];
    const bf16* Bl = Bs[buf];
    v16bf a0 = load_a_frag(Al, BK, wm, 0, lane);
    v16bf a1 = load_a_frag(Al, BK, wm + 16, 0, lane);
    v16bf b0 = load_b_frag(Bl, BK, wn, 0, lane);
    v16bf b1 = load_b_frag(Bl, BK, wn + 16, 0, lane);
    acc00 = WMMA_BF16(a0, b0, acc00);
    acc01 = WMMA_BF16(a0, b1, acc01);
    acc10 = WMMA_BF16(a1, b0, acc10);
    acc11 = WMMA_BF16(a1, b1, acc11);
    // s_wait_dscnt before the WMMAs guarantees LDS reads completed before the
    // top-of-next-iteration barrier, so buf^1 restaging can't race the reads.
  }

  // C/D layout: lane: N = lane&15 ; VGPR r: M = r + (lane>=16 ? 8 : 0)
  const int n0 = lane & 15;
  const int mh = (lane >> 4) << 3;
  const int cm = mb + wm, cn = nb + wn;
#pragma unroll
  for (int r = 0; r < 8; ++r) {
    const int row0 = cm + mh + r, row1 = cm + 16 + mh + r;
    if (cn + n0 < N) {
      C[(size_t)row0 * N + cn + n0] = acc00[r];
      C[(size_t)row1 * N + cn + n0] = acc10[r];
    }
    if (cn + 16 + n0 < N) {
      C[(size_t)row0 * N + cn + 16 + n0] = acc01[r];
      C[(size_t)row1 * N + cn + 16 + n0] = acc11[r];
    }
  }
}

// ---------------------------------------------------------------- LayerNorm
__global__ void layernorm_rows(const float* __restrict__ in, int istride, int W,
                               const float* __restrict__ gamma,
                               const float* __restrict__ beta,
                               bf16* __restrict__ out) {
  __shared__ float rs[256];
  __shared__ float rq[256];
  const int row = blockIdx.x;
  const float* x = in + (size_t)row * istride;
  float s = 0.f, sq = 0.f;
  for (int i = threadIdx.x; i < W; i += 256) { float v = x[i]; s += v; sq += v * v; }
  rs[threadIdx.x] = s; rq[threadIdx.x] = sq;
  __syncthreads();
  for (int st = 128; st > 0; st >>= 1) {
    if ((int)threadIdx.x < st) { rs[threadIdx.x] += rs[threadIdx.x + st];
                                 rq[threadIdx.x] += rq[threadIdx.x + st]; }
    __syncthreads();
  }
  const float mean = rs[0] / (float)W;
  const float var  = rq[0] / (float)W - mean * mean;
  const float inv  = rsqrtf(var + 1e-5f);
  bf16* o = out + (size_t)row * W;
  for (int i = threadIdx.x; i < W; i += 256)
    o[i] = (bf16)(gamma[i] * (x[i] - mean) * inv + beta[i]);
}

// ---------------------------------------------------------------- converters
__global__ void cvt_bf16(const float* __restrict__ in, bf16* __restrict__ out, size_t n) {
  size_t i = (size_t)blockIdx.x * 256 + threadIdx.x;
  if (i < n) out[i] = (bf16)in[i];
}
// W (K x N, f32) -> Wt (N x K, bf16)
__global__ void cvt_transpose_bf16(const float* __restrict__ in, bf16* __restrict__ out,
                                   int K, int N) {
  size_t i = (size_t)blockIdx.x * 256 + threadIdx.x;
  if (i >= (size_t)K * N) return;
  int k = (int)(i / N), n = (int)(i % N);
  out[(size_t)n * K + k] = (bf16)in[i];
}

// ---------------------------------------------------------------- RoPE helpers
__device__ __forceinline__ float rope_val(const float* base, int j, int spos) {
  int i = (j < 48) ? j : j - 48;                    // freq index (ROPE/2 = 48)
  float freq = powf(10000.0f, -(2.0f * (float)i) / (float)DH);
  float sn, cs;
  __sincosf((float)spos * freq, &sn, &cs);
  float xo = (j < 48) ? -base[j + 48] : base[j - 48]; // rotate_half
  return base[j] * cs + xo * sn;
}

// Q f32 [ROWS][3072] -> qh bf16 [B*H][S][192] with RoPE on d in [96,192)
__global__ void rope_pack_q(const float* __restrict__ Qf, bf16* __restrict__ qh) {
  size_t idx = (size_t)blockIdx.x * 256 + threadIdx.x;
  if (idx >= (size_t)ROWS * EMBED) return;
  int d = (int)(idx % DH); size_t t = idx / DH;
  int h = (int)(t % HEADS); t /= HEADS;
  int s = (int)(t % SEQ);  int b = (int)(t / SEQ);
  size_t row = (size_t)b * SEQ + s;
  const float* qp = Qf + row * EMBED + h * DH;
  float val = (d < NOPE) ? qp[d] : rope_val(qp + NOPE, d - NOPE, s);
  qh[(((size_t)b * HEADS + h) * SEQ + s) * DH + d] = (bf16)val;
}

// K_rope (last 96 cols of ckv) -> Kr bf16 [ROWS][96] with RoPE
__global__ void rope_kr(const float* __restrict__ ckv, bf16* __restrict__ Kr) {
  size_t idx = (size_t)blockIdx.x * 256 + threadIdx.x;
  if (idx >= (size_t)ROWS * ROPE) return;
  int j = (int)(idx % ROPE); size_t row = idx / ROPE;
  int s = (int)(row % SEQ);
  const float* kp = ckv + row * KVW + KVPROJ;
  Kr[row * ROPE + j] = (bf16)rope_val(kp, j, s);
}

// KV f32 [ROWS][4608] + Kr -> kh bf16 [B*H][S][192], vT bf16 [B*H][192][S]
__global__ void pack_kv(const float* __restrict__ KV, const bf16* __restrict__ Kr,
                        bf16* __restrict__ kh, bf16* __restrict__ vT) {
  size_t idx = (size_t)blockIdx.x * 256 + threadIdx.x;
  if (idx >= (size_t)ROWS * HEADS * DH) return;
  int d = (int)(idx % DH); size_t t = idx / DH;
  int s = (int)(t % SEQ); t /= SEQ;
  int h = (int)(t % HEADS); int b = (int)(t / HEADS);
  size_t row = (size_t)b * SEQ + s;
  const float* kvp = KV + row * UKVW + h * (DH + NOPE);
  size_t bh = (size_t)b * HEADS + h;
  bf16 kk = (d < NOPE) ? (bf16)kvp[d] : Kr[row * ROPE + (d - NOPE)];
  kh[(bh * SEQ + s) * DH + d] = kk;
  vT[(bh * DH + d) * SEQ + s] = (bf16)kvp[NOPE + d];
}

// ---------------------------------------------------------------- attention
// One wave per (batch*head, 16-query tile). Flash-style online softmax.
__global__ __launch_bounds__(32) void mla_attention(
    const bf16* __restrict__ qh, const bf16* __restrict__ kh,
    const bf16* __restrict__ vT, bf16* __restrict__ aout) {
  __shared__ __align__(16) bf16 Pl[16 * 32];
  const int lane = threadIdx.x;
  const int qt = blockIdx.x, bh = blockIdx.y;
  const int b = bh >> 4, h = bh & 15;
  const int qbase = qt * 16;
  const bf16* Q  = qh + (size_t)bh * SEQ * DH;   // [S][192]
  const bf16* Kt = kh + (size_t)bh * SEQ * DH;   // [S][192] == B^T for scores
  const bf16* Vt = vT + (size_t)bh * DH * SEQ;   // [192][S] == B^T for attn*V

  v16bf qa[6];
#pragma unroll
  for (int i = 0; i < 6; ++i) qa[i] = load_a_frag(Q, DH, qbase, i * 32, lane);

  v8f oacc[12];
#pragma unroll
  for (int t = 0; t < 12; ++t) oacc[t] = (v8f){0,0,0,0,0,0,0,0};
  float Mr[8], Lr[8];
#pragma unroll
  for (int r = 0; r < 8; ++r) { Mr[r] = -INFINITY; Lr[r] = 0.f; }

  const int n0 = lane & 15;
  const int mh = (lane >> 4) << 3;
  const float sc = 0.0721687836487032f;  // 1/sqrt(192)

  for (int kb = 0; kb <= qbase; kb += 32) {
    v8f s0 = {0,0,0,0,0,0,0,0}, s1 = {0,0,0,0,0,0,0,0};
#pragma unroll
    for (int i = 0; i < 6; ++i) {
      v16bf kb0 = load_b_frag(Kt, DH, kb, i * 32, lane);
      v16bf kb1 = load_b_frag(Kt, DH, kb + 16, i * 32, lane);
      s0 = WMMA_BF16(qa[i], kb0, s0);
      s1 = WMMA_BF16(qa[i], kb1, s1);
    }
#pragma unroll
    for (int r = 0; r < 8; ++r) {
      const int qg = qbase + mh + r;
      float x0 = ((kb + n0) <= qg)      ? s0[r] * sc : -INFINITY;
      float x1 = ((kb + 16 + n0) <= qg) ? s1[r] * sc : -INFINITY;
      float mx = fmaxf(x0, x1);
      for (int off = 1; off < 16; off <<= 1) mx = fmaxf(mx, __shfl_xor(mx, off, 32));
      const float newM  = fmaxf(Mr[r], mx);
      const float scale = __expf(Mr[r] - newM);
      const float p0 = __expf(x0 - newM);
      const float p1 = __expf(x1 - newM);
      float ls = p0 + p1;
      for (int off = 1; off < 16; off <<= 1) ls += __shfl_xor(ls, off, 32);
      Lr[r] = Lr[r] * scale + ls;
      Mr[r] = newM;
#pragma unroll
      for (int t = 0; t < 12; ++t) oacc[t][r] *= scale;
      Pl[(mh + r) * 32 + n0]      = (bf16)p0;
      Pl[(mh + r) * 32 + 16 + n0] = (bf16)p1;
    }
    __syncthreads();
    v16bf pa = load_a_frag(Pl, 32, 0, 0, lane);
#pragma unroll
    for (int t = 0; t < 12; ++t) {
      v16bf vb = load_b_frag(Vt, SEQ, t * 16, kb, lane);
      oacc[t] = WMMA_BF16(pa, vb, oacc[t]);
    }
    __syncthreads();
  }

  // normalize + write to attn_out layout [b][s][h*192+d]
#pragma unroll
  for (int r = 0; r < 8; ++r) {
    const float inv = 1.f / Lr[r];
    size_t row = (size_t)b * SEQ + qbase + mh + r;
#pragma unroll
    for (int t = 0; t < 12; ++t)
      aout[row * EMBED + h * DH + t * 16 + n0] = (bf16)(oacc[t][r] * inv);
  }
}

// ---------------------------------------------------------------- workspace arena
static constexpr size_t SZ_XB   = (size_t)ROWS * EMBED * 2;     // x bf16
static constexpr size_t SZ_WDQ  = (size_t)QPROJ * EMBED * 2;    // W_dq^T
static constexpr size_t SZ_WUQ  = (size_t)EMBED * QPROJ * 2;    // W_uq^T
static constexpr size_t SZ_WDKV = (size_t)KVW_PAD * EMBED * 2;  // W_dkv^T (N-padded)
static constexpr size_t SZ_WUKV = (size_t)UKVW * KVPROJ * 2;    // W_ukv^T
static constexpr size_t SZ_WO   = (size_t)EMBED * EMBED * 2;    // W_o (as-is == B^T)
static constexpr size_t SZ_QH   = (size_t)ROWS * EMBED * 2;
static constexpr size_t SZ_KH   = (size_t)ROWS * EMBED * 2;
static constexpr size_t SZ_VT   = (size_t)ROWS * EMBED * 2;
static constexpr size_t SZ_KR   = (size_t)ROWS * ROPE * 2;
static constexpr size_t SZ_AO   = (size_t)ROWS * EMBED * 2;
static constexpr size_t SZ_TA   = (size_t)ROWS * UKVW * 4;      // biggest f32 temp (KV)
// offsets
static constexpr size_t O_XB   = 0;
static constexpr size_t O_WDQ  = O_XB + SZ_XB;
static constexpr size_t O_WUQ  = O_WDQ + SZ_WDQ;
static constexpr size_t O_WDKV = O_WUQ + SZ_WUQ;
static constexpr size_t O_WUKV = O_WDKV + SZ_WDKV;
static constexpr size_t O_WO   = O_WUKV + SZ_WUKV;
static constexpr size_t O_QH   = O_WO + SZ_WO;
static constexpr size_t O_KH   = O_QH + SZ_QH;
static constexpr size_t O_VT   = O_KH + SZ_KH;
static constexpr size_t O_KR   = O_VT + SZ_VT;
static constexpr size_t O_AO   = O_KR + SZ_KR;
static constexpr size_t O_TA   = O_AO + SZ_AO;   // f32 temps: cq_pre -> Q -> ckv -> KV
static constexpr size_t O_TB   = O_TA + SZ_TA;   // bf16 temps: cq_b -> kv_lora_b

extern "C" void kernel_launch(void* const* d_in, const int* in_sizes, int n_in,
                              void* d_out, int out_size, void* d_ws, size_t ws_size,
                              hipStream_t stream) {
  (void)in_sizes; (void)n_in; (void)out_size; (void)ws_size;
  const float* x      = (const float*)d_in[0];
  const float* W_dq   = (const float*)d_in[1];
  const float* W_uq   = (const float*)d_in[2];
  const float* q_g    = (const float*)d_in[3];
  const float* q_be   = (const float*)d_in[4];
  const float* W_dkv  = (const float*)d_in[5];
  const float* W_ukv  = (const float*)d_in[6];
  const float* kv_g   = (const float*)d_in[7];
  const float* kv_be  = (const float*)d_in[8];
  const float* W_o    = (const float*)d_in[9];
  float* out = (float*)d_out;

  char* w = (char*)d_ws;
  bf16*  Xb    = (bf16*)(w + O_XB);
  bf16*  Wdq_t = (bf16*)(w + O_WDQ);
  bf16*  Wuq_t = (bf16*)(w + O_WUQ);
  bf16*  Wdkv_t= (bf16*)(w + O_WDKV);
  bf16*  Wukv_t= (bf16*)(w + O_WUKV);
  bf16*  Wo_b  = (bf16*)(w + O_WO);
  bf16*  qhB   = (bf16*)(w + O_QH);
  bf16*  khB   = (bf16*)(w + O_KH);
  bf16*  vTB   = (bf16*)(w + O_VT);
  bf16*  KrB   = (bf16*)(w + O_KR);
  bf16*  aoB   = (bf16*)(w + O_AO);
  float* TAf   = (float*)(w + O_TA);
  bf16*  TBb   = (bf16*)(w + O_TB);

  const dim3 blk256(256);
  const dim3 blkG(128);
  auto gr = [](size_t n) { return dim3((unsigned)((n + 255) / 256)); };

  // 1) bf16 staging of activations + weights (weights transposed into B^T layout)
  cvt_bf16<<<gr((size_t)ROWS * EMBED), blk256, 0, stream>>>(x, Xb, (size_t)ROWS * EMBED);
  cvt_transpose_bf16<<<gr((size_t)EMBED * QPROJ), blk256, 0, stream>>>(W_dq, Wdq_t, EMBED, QPROJ);
  cvt_transpose_bf16<<<gr((size_t)QPROJ * EMBED), blk256, 0, stream>>>(W_uq, Wuq_t, QPROJ, EMBED);
  cvt_transpose_bf16<<<gr((size_t)EMBED * KVW), blk256, 0, stream>>>(W_dkv, Wdkv_t, EMBED, KVW);
  cvt_transpose_bf16<<<gr((size_t)KVPROJ * UKVW), blk256, 0, stream>>>(W_ukv, Wukv_t, KVPROJ, UKVW);
  cvt_bf16<<<gr((size_t)EMBED * EMBED), blk256, 0, stream>>>(W_o, Wo_b, (size_t)EMBED * EMBED);

  // 2) cq_pre = x @ W_dq  (f32 in TA), then LN -> bf16 (TB)
  gemm_bf16_wmma<<<dim3(QPROJ / 64, ROWS / 64), blkG, 0, stream>>>(
      Xb, Wdq_t, TAf, ROWS, QPROJ, EMBED);
  layernorm_rows<<<dim3(ROWS), blk256, 0, stream>>>(TAf, QPROJ, QPROJ, q_g, q_be, TBb);

  // 3) Q = cq @ W_uq (f32 in TA, overwrites cq_pre), RoPE + pack -> qh
  gemm_bf16_wmma<<<dim3(EMBED / 64, ROWS / 64), blkG, 0, stream>>>(
      TBb, Wuq_t, TAf, ROWS, EMBED, QPROJ);
  rope_pack_q<<<gr((size_t)ROWS * EMBED), blk256, 0, stream>>>(TAf, qhB);

  // 4) ckv = x @ W_dkv (f32 in TA), LN(first 2048) -> TB, RoPE(last 96) -> Kr
  gemm_bf16_wmma<<<dim3(KVW_PAD / 64, ROWS / 64), blkG, 0, stream>>>(
      Xb, Wdkv_t, TAf, ROWS, KVW, EMBED);
  layernorm_rows<<<dim3(ROWS), blk256, 0, stream>>>(TAf, KVW, KVPROJ, kv_g, kv_be, TBb);
  rope_kr<<<gr((size_t)ROWS * ROPE), blk256, 0, stream>>>(TAf, KrB);

  // 5) KV = kv_lora @ W_ukv (f32 in TA), pack -> kh, vT
  gemm_bf16_wmma<<<dim3(UKVW / 64, ROWS / 64), blkG, 0, stream>>>(
      TBb, Wukv_t, TAf, ROWS, UKVW, KVPROJ);
  pack_kv<<<gr((size_t)ROWS * HEADS * DH), blk256, 0, stream>>>(TAf, KrB, khB, vTB);

  // 6) causal flash attention -> attn_out bf16 [b][s][h*192+d]
  mla_attention<<<dim3(SEQ / 16, BATCH * HEADS), dim3(32), 0, stream>>>(qhB, khB, vTB, aoB);

  // 7) final projection: out = attn_out @ W_o^T  (B^T == W_o as stored)
  gemm_bf16_wmma<<<dim3(EMBED / 64, ROWS / 64), blkG, 0, stream>>>(
      aoB, Wo_b, out, ROWS, EMBED, EMBED);
}
